// LMFELayer_72249939853402
// MI455X (gfx1250) — compile-verified, matrix operand order checked by reading, not run on hardware
//
#include <hip/hip_runtime.h>

typedef __attribute__((ext_vector_type(16))) _Float16 v16h;
typedef __attribute__((ext_vector_type(8)))  float    v8f;
typedef __attribute__((ext_vector_type(4)))  float    v4f;

namespace {
constexpr int kFFT         = 512;
constexpr int kHalf        = 257;
constexpr int kMels        = 64;
constexpr int kKsteps      = 9;               // ceil(257/32) -> K padded to 288 (zero filters)
constexpr int kNTiles      = 4;               // 64 mel bins = 4 x 16-wide WMMA tiles
constexpr int kRowsPerWave = 16;
constexpr int kWavesPerWG  = 8;
constexpr int kRowsPerWG   = kRowsPerWave * kWavesPerWG;   // 128
constexpr int kLdsHalves   = kNTiles * kKsteps * 32 * 16;  // 18432 f16 = 36 KB
constexpr int kTotalRows   = 256 * 256;                    // B * NUM_FRAMES
}

__global__ __launch_bounds__(256)
void lmfe_wmma_kernel(const float* __restrict__ x,
                      const float* __restrict__ fb,
                      float* __restrict__ out)
{
    // Filter bank staged in LDS as ready-to-use WMMA B fragments:
    // index = ((ntile * kKsteps + s) * 32 + lane) * 16 + half
    __shared__ __align__(32) _Float16 ldsB[kLdsHalves];

    const int tid = threadIdx.x;

    // --- Stage fb (f32, 64x257) -> LDS f16 B fragments, zero-padded K ---
    // 16-bit B operand striping (ISA 7.12.2): lane L<16 holds column N=L with
    // halves {K+0..7, K+16..23}; lane L+16 holds halves {K+8..15, K+24..31}.
    for (int idx = tid; idx < kLdsHalves; idx += 256) {
        const int h    = idx & 15;
        const int lane = (idx >> 4) & 31;
        const int s    = (idx >> 9) % kKsteps;
        const int t    = idx / (kKsteps * 512);
        const int n    = t * 16 + (lane & 15);
        const int koff = (h & 7) + 8 * (((h >> 3) << 1) + (lane >> 4));
        const int k    = s * 32 + koff;
        const float v  = (k < kHalf) ? fb[n * kHalf + k] : 0.0f;
        ldsB[idx] = (_Float16)v;
    }
    __syncthreads();

    const int  lane  = tid & 31;
    const int  wave  = tid >> 5;
    const long mbase = (long)blockIdx.x * kRowsPerWG + (long)wave * kRowsPerWave;

    // A operand: lane L and L+16 both hold row M = mbase + (L&15); L<16 takes
    // K halves {0..7,16..23}, L>=16 takes {8..15,24..31} (kgrp offset of 8).
    const float* px   = x + (mbase + (lane & 15)) * kFFT;
    const int    kgrp = (lane >> 4) * 8;

    v8f acc[kNTiles] = {};

    for (int s = 0; s < kKsteps; ++s) {
        const int kA = s * 32 + kgrp;   // halves 0..7  (two b128 loads)
        const int kB = kA + 16;         // halves 8..15 (two b128 loads)
        const v4f a0 = *(const v4f*)(px + kA);
        const v4f a1 = *(const v4f*)(px + kA + 4);
        const v4f a2 = *(const v4f*)(px + kB);
        const v4f a3 = *(const v4f*)(px + kB + 4);

        // mag = (x + 1)^2 computed in f32, narrowed to f16 for the WMMA.
        v16h af;
#pragma unroll
        for (int i = 0; i < 4; ++i) { const float v = a0[i] + 1.0f; af[i]      = (_Float16)(v * v); }
#pragma unroll
        for (int i = 0; i < 4; ++i) { const float v = a1[i] + 1.0f; af[4 + i]  = (_Float16)(v * v); }
#pragma unroll
        for (int i = 0; i < 4; ++i) { const float v = a2[i] + 1.0f; af[8 + i]  = (_Float16)(v * v); }
#pragma unroll
        for (int i = 0; i < 4; ++i) { const float v = a3[i] + 1.0f; af[12 + i] = (_Float16)(v * v); }

#pragma unroll
        for (int t = 0; t < kNTiles; ++t) {
            const v16h bf = *(const v16h*)(&ldsB[((t * kKsteps + s) * 32 + lane) * 16]);
            // D = A x B + C : v_wmma_f32_16x16x32_f16
            acc[t] = __builtin_amdgcn_wmma_f32_16x16x32_f16(
                /*neg_a=*/false, af, /*neg_b=*/false, bf,
                /*c_mod=*/(short)0, acc[t], /*reuse_a=*/false, /*reuse_b=*/false);
        }
    }

    // --- Epilogue: floor(log2(.)) and store. C layout: VGPR r holds rows
    // M = r (lanes 0-15) / r+8 (lanes 16-31), N = ntile*16 + (lane&15). ---
#pragma unroll
    for (int t = 0; t < kNTiles; ++t) {
#pragma unroll
        for (int r = 0; r < 8; ++r) {
            const long m = mbase + r + (lane >> 4) * 8;
            const int  n = t * 16 + (lane & 15);
            out[m * kMels + n] = floorf(__log2f(acc[t][r]));
        }
    }
}

extern "C" void kernel_launch(void* const* d_in, const int* in_sizes, int n_in,
                              void* d_out, int out_size, void* d_ws, size_t ws_size,
                              hipStream_t stream) {
    const float* x  = (const float*)d_in[0];   // (256, 256, 512, 1) f32
    const float* fb = (const float*)d_in[1];   // (64, 257) f32
    float* out      = (float*)d_out;           // (256, 256, 64, 1) f32

    const dim3 grid(kTotalRows / kRowsPerWG);  // 512 workgroups
    const dim3 block(256);                     // 8 wave32s
    lmfe_wmma_kernel<<<grid, block, 0, stream>>>(x, fb, out);
}
